// GCN_13606456393732
// MI455X (gfx1250) — matrix operand order repaired and checked by reading, not compiled, not run on hardware
//
#include <hip/hip_runtime.h>

// ---------------------------------------------------------------------------
// 2-layer GCN for MI455X (gfx1250, wave32, WMMA).
//
//   support  = x @ w1                          (bf16 WMMA, f32 accum)
//   h        = spmm(adj, support)              (f32 atomics, L2-resident gather)
//   support2 = relu(h + b1) @ w2               (epilogue fused into WMMA A-load)
//   out      = spmm(adj, support2) + b2        (out pre-seeded with b2)
//
// N=100000, F_IN=512, H=256, C=64, E=3.2M
//
// GEMM register blocking: one wave owns a 32(M) x 64(N) strip. Per K-step:
// 2 A fragments (f32 load + cvt + optional bias/relu) and 4 B fragments are
// built into INDEPENDENT registers, then 8 v_wmma_f32_16x16x32_bf16 issues
// drain them -> loads clause together and overlap with matrix issue.
// ---------------------------------------------------------------------------

#define GCN_N    100000
#define GCN_FIN  512
#define GCN_H    256
#define GCN_C    64

typedef __attribute__((ext_vector_type(16))) __bf16 v16bf;
typedef __attribute__((ext_vector_type(8)))  __bf16 v8bf;
typedef __attribute__((ext_vector_type(8)))  float  v8f;
typedef __attribute__((ext_vector_type(4)))  float  v4f;

// A fragment: CDNA5 16-bit A 16x32 layout. Lane(half,r) holds row m=r,
// K = {k0+half*8 ..+7} and {k0+16+half*8 ..+7}; f32 -> bf16 in-register,
// optionally fused with bias+relu (layer-2 epilogue).
__device__ __forceinline__ v16bf gcn_a_frag(const float* __restrict__ arow,
                                            const float* __restrict__ bias,
                                            int k0, int half, int relu_bias)
{
    union { v16bf v; __bf16 e[16]; } a;
    const float* a0 = arow + k0 + half * 8;
    const float* a1 = arow + k0 + 16 + half * 8;
    if (relu_bias) {
        const float* c0 = bias + k0 + half * 8;
        const float* c1 = bias + k0 + 16 + half * 8;
        #pragma unroll
        for (int i = 0; i < 8; ++i) {
            a.e[i]     = (__bf16)__builtin_fmaxf(a0[i] + c0[i], 0.0f);
            a.e[8 + i] = (__bf16)__builtin_fmaxf(a1[i] + c1[i], 0.0f);
        }
    } else {
        v4f x0 = *(const v4f*)(a0);
        v4f x1 = *(const v4f*)(a0 + 4);
        v4f x2 = *(const v4f*)(a1);
        v4f x3 = *(const v4f*)(a1 + 4);
        #pragma unroll
        for (int i = 0; i < 4; ++i) {
            a.e[i]      = (__bf16)x0[i];
            a.e[4 + i]  = (__bf16)x1[i];
            a.e[8 + i]  = (__bf16)x2[i];
            a.e[12 + i] = (__bf16)x3[i];
        }
    }
    return a.v;
}

// B fragment: lane(half,r) holds column n, K = {k0+half*8 ..+7} and
// {k0+16+half*8 ..+7} (contiguous runs in Bt because Bt is B transposed).
__device__ __forceinline__ v16bf gcn_b_frag(const __bf16* __restrict__ brow,
                                            int k0, int half)
{
    union { v16bf v; v8bf h2[2]; } b;
    const __bf16* bp = brow + k0 + half * 8;
    b.h2[0] = *(const v8bf*)(bp);
    b.h2[1] = *(const v8bf*)(bp + 16);
    return b.v;
}

// ---------------------------------------------------------------------------
// bf16-WMMA GEMM:  D[M,Nn] = cvt_bf16(epilogue(A)) @ Bt^T
//   A   : f32 [M, K] row-major    Bt: bf16 [Nn, K] row-major
//   bias: f32 [K]; if relu_bias!=0, A-fragment = relu(A + bias) before cvt
// One wave computes a 32x64 strip (8 accumulators); 8 waves per block.
// ---------------------------------------------------------------------------
__global__ __launch_bounds__(256) void gcn_gemm_bf16_wmma(
    const float* __restrict__ A,
    const __bf16* __restrict__ Bt,
    const float* __restrict__ bias,
    float* __restrict__ D,
    int M, int Nn, int K, int relu_bias)
{
    const int lane = threadIdx.x & 31;
    const int wave = threadIdx.x >> 5;
    const int ngroups = Nn >> 6;                   // 64-wide N strips
    const int mtiles  = M >> 5;                    // 32-row M strips

    int tile = blockIdx.x * 8 + wave;
    if (tile >= mtiles * ngroups) return;          // wave-uniform; EXEC all-1s
    const int mt = tile / ngroups;
    const int ng = tile - mt * ngroups;

    const int half = lane >> 4;                    // 0: lanes 0-15, 1: 16-31
    const int r    = lane & 15;

    const float* arow0 = A + (size_t)(mt * 32 + r) * K;
    const float* arow1 = A + (size_t)(mt * 32 + 16 + r) * K;

    const __bf16* brow[4];
    #pragma unroll
    for (int j = 0; j < 4; ++j)
        brow[j] = Bt + (size_t)(ng * 64 + j * 16 + r) * K;

    v8f acc[2][4] = {};

    for (int k0 = 0; k0 < K; k0 += 32) {
        // build both A fragments and all 4 B fragments first (independent
        // registers -> loads clause and overlap with the WMMA burst)
        v16bf a0 = gcn_a_frag(arow0, bias, k0, half, relu_bias);
        v16bf a1 = gcn_a_frag(arow1, bias, k0, half, relu_bias);

        v16bf bf[4];
        #pragma unroll
        for (int j = 0; j < 4; ++j)
            bf[j] = gcn_b_frag(brow[j], k0, half);

        #pragma unroll
        for (int j = 0; j < 4; ++j) {
            acc[0][j] = __builtin_amdgcn_wmma_f32_16x16x32_bf16(
                            false, a0, false, bf[j], (short)0, acc[0][j], false, false);
            acc[1][j] = __builtin_amdgcn_wmma_f32_16x16x32_bf16(
                            false, a1, false, bf[j], (short)0, acc[1][j], false, false);
        }
    }

    // store: VGPR v of lane(half,r) = element (m = base + half*8 + v, n)
    #pragma unroll
    for (int mi = 0; mi < 2; ++mi) {
        float* dbase = D + (size_t)(mt * 32 + mi * 16 + half * 8) * Nn
                         + ng * 64 + r;
        #pragma unroll
        for (int v = 0; v < 8; ++v) {
            size_t ro = (size_t)v * Nn;
            dbase[ro +  0] = acc[mi][0][v];
            dbase[ro + 16] = acc[mi][1][v];
            dbase[ro + 32] = acc[mi][2][v];
            dbase[ro + 48] = acc[mi][3][v];
        }
    }
}

// ---------------------------------------------------------------------------
// COO SpMM via f32 atomics: out[row] += val * dense[col], one wave per edge.
// Dense rows are L2-resident (support: 100MB, support2: 25.6MB vs 192MB L2).
// ---------------------------------------------------------------------------
template <int DD>
__global__ __launch_bounds__(256) void gcn_spmm_atomic(
    const int* __restrict__ rows, const int* __restrict__ cols,
    const float* __restrict__ vals,
    const float* __restrict__ dense,
    float* __restrict__ out, int E)
{
    const int lane = threadIdx.x & 31;
    const int wave = threadIdx.x >> 5;
    int e = blockIdx.x * 8 + wave;
    if (e >= E) return;

    // prefetch next edge's source row (global_prefetch_b8)
    if (e + 8 < E) {
        int cn = cols[e + 8];
        __builtin_prefetch(dense + (size_t)cn * DD, 0, 0);
    }

    const int   rr = rows[e];
    const int   cc = cols[e];
    const float vv = vals[e];
    const float* __restrict__ src = dense + (size_t)cc * DD;
    float* __restrict__ dst = out + (size_t)rr * DD;

    #pragma unroll
    for (int i = lane; i < DD; i += 32)            // 128B coalesced per step
        atomicAdd(dst + i, vv * src[i]);
}

// ---------------------------------------------------------------------------
// Helpers: weight transpose+cvt, zero fill, bias broadcast init.
// ---------------------------------------------------------------------------
__global__ void gcn_transpose_cvt_bf16(const float* __restrict__ src,
                                       __bf16* __restrict__ dst, int K, int Nn)
{
    int i = blockIdx.x * blockDim.x + threadIdx.x;   // src [K][Nn] -> dst [Nn][K]
    if (i < K * Nn) {
        int k = i / Nn, n = i - k * Nn;
        dst[(size_t)n * K + k] = (__bf16)src[i];
    }
}

__global__ void gcn_zero_f32(float* __restrict__ p, int n)
{
    int i = blockIdx.x * blockDim.x + threadIdx.x;
    if (i < n) p[i] = 0.0f;
}

__global__ void gcn_init_bias_rows(float* __restrict__ out,
                                   const float* __restrict__ bias, int total)
{
    int i = blockIdx.x * blockDim.x + threadIdx.x;
    if (i < total) out[i] = bias[i & (GCN_C - 1)];   // GCN_C = 64, power of two
}

// ---------------------------------------------------------------------------
// Launcher
// ---------------------------------------------------------------------------
extern "C" void kernel_launch(void* const* d_in, const int* in_sizes, int n_in,
                              void* d_out, int out_size, void* d_ws, size_t ws_size,
                              hipStream_t stream)
{
    const float* x       = (const float*)d_in[0];   // [N, F_IN]
    const int*   adj_row = (const int*)  d_in[1];   // [E]
    const int*   adj_col = (const int*)  d_in[2];   // [E]
    const float* adj_val = (const float*)d_in[3];   // [E]
    const float* w1      = (const float*)d_in[4];   // [F_IN, H]
    const float* b1      = (const float*)d_in[5];   // [H]
    const float* w2      = (const float*)d_in[6];   // [H, C]
    const float* b2      = (const float*)d_in[7];   // [C]
    float*       out     = (float*)d_out;           // [N, C]

    const int E = in_sizes[1];

    // ---- workspace carve (all 256B-aligned, ~231 MB total) ----
    char*  ws  = (char*)d_ws;
    size_t off = 0;
    auto carve = [&](size_t bytes) {
        void* p = ws + off;
        off += (bytes + 255) & ~(size_t)255;
        return p;
    };
    float*  support  = (float*) carve((size_t)GCN_N * GCN_H * 4);   // 102.4 MB
    float*  h        = (float*) carve((size_t)GCN_N * GCN_H * 4);   // 102.4 MB
    float*  support2 = (float*) carve((size_t)GCN_N * GCN_C * 4);   //  25.6 MB
    __bf16* w1t      = (__bf16*)carve((size_t)GCN_FIN * GCN_H * 2); //  0.26 MB
    __bf16* w2t      = (__bf16*)carve((size_t)GCN_H * GCN_C * 2);   //  0.03 MB

    // ---- 1. weight prep: transpose + cvt to bf16 ----
    gcn_transpose_cvt_bf16<<<(GCN_FIN * GCN_H + 255) / 256, 256, 0, stream>>>(
        w1, w1t, GCN_FIN, GCN_H);
    gcn_transpose_cvt_bf16<<<(GCN_H * GCN_C + 255) / 256, 256, 0, stream>>>(
        w2, w2t, GCN_H, GCN_C);

    // ---- 2. support = x @ w1  (3125 m-strips x 4 n-groups = 12500 waves) ----
    {
        int tiles  = (GCN_N / 32) * (GCN_H / 64);
        int blocks = (tiles + 7) / 8;
        gcn_gemm_bf16_wmma<<<blocks, 256, 0, stream>>>(
            x, w1t, nullptr, support, GCN_N, GCN_H, GCN_FIN, 0);
    }

    // ---- 3. h = spmm(adj, support) ----
    gcn_zero_f32<<<(GCN_N * GCN_H + 255) / 256, 256, 0, stream>>>(
        h, GCN_N * GCN_H);
    gcn_spmm_atomic<GCN_H><<<(E + 7) / 8, 256, 0, stream>>>(
        adj_row, adj_col, adj_val, support, h, E);

    // ---- 4. support2 = relu(h + b1) @ w2  (epilogue fused into A-load) ----
    {
        int tiles  = (GCN_N / 32) * (GCN_C / 64);
        int blocks = (tiles + 7) / 8;
        gcn_gemm_bf16_wmma<<<blocks, 256, 0, stream>>>(
            h, w2t, b1, support2, GCN_N, GCN_C, GCN_H, 1);
    }

    // ---- 5. out = spmm(adj, support2) + b2 ----
    gcn_init_bias_rows<<<(GCN_N * GCN_C + 255) / 256, 256, 0, stream>>>(
        out, b2, GCN_N * GCN_C);
    gcn_spmm_atomic<GCN_C><<<(E + 7) / 8, 256, 0, stream>>>(
        adj_row, adj_col, adj_val, support2, out, E);
}
